// FNOTransferHead_7550552506854
// MI455X (gfx1250) — compile-verified, matrix operand order checked by reading, not run on hardware
//
#include <hip/hip_runtime.h>
#include <hip/hip_bf16.h>
#include <math.h>

#define BATCH 128
#define WIDTH 64
#define LSEQ  4096
#define MODES 32
#define DLAT  128
#define LIFTL 64
#define PROJ  128
#define KSPLIT 8

typedef __attribute__((ext_vector_type(16))) __bf16 v16bf;
typedef __attribute__((ext_vector_type(8)))  __bf16 v8bf;
typedef __attribute__((ext_vector_type(8)))  float  v8f;
typedef unsigned int u32x4 __attribute__((ext_vector_type(4)));
typedef int i32x4 __attribute__((ext_vector_type(4)));
typedef int i32x8 __attribute__((ext_vector_type(8)));

#if __has_builtin(__builtin_amdgcn_tensor_load_to_lds)
#define TDM_AVAILABLE 1
#else
#define TDM_AVAILABLE 0
#endif

__device__ __forceinline__ unsigned short f2bf(float f) {
  unsigned int u = __float_as_uint(f);
  u += 0x7FFFu + ((u >> 16) & 1u);          // round-to-nearest-even
  return (unsigned short)(u >> 16);
}
__device__ __forceinline__ float bf2f(unsigned short h) {
  return __uint_as_float(((unsigned int)h) << 16);
}
__device__ __forceinline__ float gelu_exact(float x) {
  return 0.5f * x * (1.0f + erff(x * 0.70710678118654752f));
}

// Fragment loader for 16-bit A/B operands of v_wmma_f32_16x16x32_bf16.
// Lane L (0-15): K = k0+[0..7] and k0+16+[0..7];  lane L (16-31): +8 offset.
// Works for global or LDS-backed pointers (addrspace inferred -> ds_load_b128).
__device__ __forceinline__ v16bf load_frag(const unsigned short* rowbase, int k0, int lane) {
  int koff = ((lane >> 4) & 1) << 3;
  v8bf lo = *reinterpret_cast<const v8bf*>(rowbase + k0 + koff);
  v8bf hi = *reinterpret_cast<const v8bf*>(rowbase + k0 + 16 + koff);
  v16bf r;
#pragma unroll
  for (int i = 0; i < 8; ++i) { r[i] = lo[i]; r[8 + i] = hi[i]; }
  return r;
}

#define WMMA_BF16(a, b, c) \
  __builtin_amdgcn_wmma_f32_16x16x32_bf16(false, (a), false, (b), (short)0, (c), false, false)

// gfx1250 async copy: 16 bytes per active lane, global -> LDS, tracked by ASYNCcnt.
// LDS operand wants the 32-bit LDS offset; for the LDS aperture the flat
// address's low 32 bits ARE the LDS offset (ISA 10.2), so truncation is exact.
__device__ __forceinline__ void async_copy16(const void* gsrc, void* ldst) {
  unsigned lds = (unsigned)(size_t)ldst;
  unsigned long long ga = (unsigned long long)(size_t)gsrc;
  asm volatile("global_load_async_to_lds_b128 %0, %1, off"
               :: "v"(lds), "v"(ga) : "memory");
}
__device__ __forceinline__ void async_wait_all() {
  asm volatile("s_wait_asynccnt 0x0" ::: "memory");
}

#if TDM_AVAILABLE
// Tensor Data Mover: 1D contiguous tile copy (ndwords 32-bit words), one
// instruction per call, tracked by TENSORcnt. D# bit layout per ISA 8.3/8.4:
//   group0: [0]=count=1, [1]=lds_addr, [2]=ga[31:0], [3]=ga[56:32]|type=2<<30
//   group1: data_size=4B(2<<16); tensor_dim0/tile_dim0 = ndwords; dim1 = 1;
//           tensor_dim0_stride = ndwords; no multicast/pad/iterate/barrier.
__device__ __forceinline__ void tdm_copy_dwords(const void* gsrc, void* ldst,
                                                unsigned ndwords) {
  unsigned long long ga = (unsigned long long)(size_t)gsrc;
  unsigned la = (unsigned)(size_t)ldst;
  u32x4 g0;
  g0[0] = 1u;
  g0[1] = la;
  g0[2] = (unsigned)(ga & 0xFFFFFFFFu);
  g0[3] = (unsigned)((ga >> 32) & 0x01FFFFFFu) | (2u << 30);
  i32x8 g1;
  g1[0] = (int)(2u << 16);                         // data_size = 4 bytes
  g1[1] = (int)((ndwords & 0xFFFFu) << 16);        // tensor_dim0 lo16
  g1[2] = (int)((ndwords >> 16) | (1u << 16));     // tensor_dim0 hi16, tensor_dim1=1
  g1[3] = (int)((ndwords & 0xFFFFu) << 16);        // tile_dim0
  g1[4] = 0;                                       // tile_dim1/2 unused
  g1[5] = (int)ndwords;                            // tensor_dim0_stride lo32
  g1[6] = 0;
  g1[7] = 0;
  i32x4 gz = {0, 0, 0, 0};
#if defined(__clang_major__) && (__clang_major__ >= 23)
  i32x8 gz8 = {0, 0, 0, 0, 0, 0, 0, 0};
  __builtin_amdgcn_tensor_load_to_lds(g0, g1, gz, gz, gz8, 0);
#else
  __builtin_amdgcn_tensor_load_to_lds(g0, g1, gz, gz, 0);
#endif
}
#endif

// ---------------------------------------------------------------- basis tables
// T  : (64, 4096) row-major   rows 0..31 = cos(2*pi*m*l/L), rows 32..63 = -sin
// Tt : (4096, 64) row-major   transposed copy (K-contiguous for hmat B operand)
__global__ void k_basis(unsigned short* T, unsigned short* Tt) {
  int idx = blockIdx.x * blockDim.x + threadIdx.x;
  if (idx >= WIDTH * LSEQ) return;
  int n = idx >> 12, l = idx & (LSEQ - 1);
  int m = n & 31;
  int ph = (m * l) & (LSEQ - 1);                 // exact phase mod L
  float ang = 6.28318530717958647692f * (float)ph / (float)LSEQ;
  float v = (n < MODES) ? cosf(ang) : -sinf(ang);
  unsigned short bv = f2bf(v);
  T[n * LSEQ + l] = bv;
  Tt[l * WIDTH + n] = bv;
}

// ---------------------------------------------------------------- conversions
__global__ void k_cvt(const float* src, unsigned short* dst, int n) {
  int i = blockIdx.x * blockDim.x + threadIdx.x;
  if (i < n) dst[i] = f2bf(src[i]);
}
// lift_w (128 x 4096) -> lwT (4096 x 128) bf16 (K-contiguous columns)
__global__ void k_cvt_liftw(const float* src, unsigned short* dst) {
  int i = blockIdx.x * blockDim.x + threadIdx.x;
  if (i >= DLAT * (WIDTH * LIFTL)) return;
  int k = i >> 12, j = i & 4095;
  dst[j * DLAT + k] = f2bf(src[i]);
}

// ---------------------------------------------------------------- lift GEMM
// u(128 x 4096) = z(128 x 128) @ lift_w(128 x 4096) + lift_b   (WMMA, M=batch)
// B tile (64 cols x K=128) is identical for all 8 waves -> stage via async LDS.
__global__ void k_lift(const unsigned short* zbf, const unsigned short* lwT,
                       const float* lift_b, float* u) {
  __shared__ unsigned short ldsB[64 * DLAT];    // 16 KB
  int tid = threadIdx.x;
  int lane = tid & 31, wave = tid >> 5;
  int m0 = wave * 16;
  int n0 = blockIdx.x * 64;
  const char* gB = (const char*)(lwT + (size_t)n0 * DLAT);
#pragma unroll
  for (int it = 0; it < 4; ++it) {
    unsigned off = (unsigned)tid * 16u + (unsigned)it * 4096u;
    async_copy16(gB + off, (char*)ldsB + off);
  }
  async_wait_all();
  __syncthreads();

  const unsigned short* arow = zbf + (m0 + (lane & 15)) * DLAT;
  v8f acc[4] = {};
#pragma unroll
  for (int k0 = 0; k0 < DLAT; k0 += 32) {
    v16bf a = load_frag(arow, k0, lane);
#pragma unroll
    for (int s = 0; s < 4; ++s) {
      const unsigned short* brow = ldsB + (s * 16 + (lane & 15)) * DLAT;
      v16bf b = load_frag(brow, k0, lane);
      acc[s] = WMMA_BF16(a, b, acc[s]);
    }
  }
#pragma unroll
  for (int s = 0; s < 4; ++s) {
    int col = n0 + s * 16 + (lane & 15);
    float bias = lift_b[col];                   // hoisted: one load per tile col
#pragma unroll
    for (int j = 0; j < 8; ++j) {
      int row = m0 + j + ((lane >> 4) << 3);
      u[row * (WIDTH * LIFTL) + col] = acc[s][j] + bias;
    }
  }
}

// ---------------------------------------------------------------- cond MLP (tiny)
__global__ void k_cond(const float* ip, const float* w1, const float* b1,
                       const float* w2, const float* b2, float* cond) {
  int b = blockIdx.x * blockDim.x + threadIdx.x;
  if (b >= BATCH) return;
  float g[WIDTH];
  for (int j = 0; j < WIDTH; ++j) {
    float s = b1[j];
    for (int i = 0; i < 8; ++i) s += ip[b * 8 + i] * w1[i * WIDTH + j];
    g[j] = gelu_exact(s);
  }
  for (int c = 0; c < WIDTH; ++c) {
    float s = b2[c];
    for (int j = 0; j < WIDTH; ++j) s += g[j] * w2[j * WIDTH + c];
    cond[b * WIDTH + c] = s;
  }
}

// ---------------------------------------------------------------- interp + cond add
__global__ void k_interp(const float* u, const float* cond,
                         unsigned short* x_cl, unsigned short* x_lc) {
  size_t idx = (size_t)blockIdx.x * blockDim.x + threadIdx.x;
  if (idx >= (size_t)BATCH * WIDTH * LSEQ) return;
  int l = (int)(idx & (LSEQ - 1));
  int c = (int)((idx >> 12) & (WIDTH - 1));
  int b = (int)(idx >> 18);
  float src = (l + 0.5f) * ((float)LIFTL / (float)LSEQ) - 0.5f;
  src = fminf(fmaxf(src, 0.0f), (float)(LIFTL - 1));
  int i0 = (int)src;
  int i1 = min(i0 + 1, LIFTL - 1);
  float w = src - (float)i0;
  const float* ub = u + b * (WIDTH * LIFTL) + c * LIFTL;
  float val = ub[i0] * (1.0f - w) + ub[i1] * w + cond[b * WIDTH + c];
  unsigned short bv = f2bf(val);
  x_cl[idx] = bv;
  x_lc[((size_t)b * LSEQ + l) * WIDTH + c] = bv;
}

// ---------------------------------------------------------------- forward DFT (WMMA)
// K=4096 split into KSPLIT slices for occupancy; partial F summed by k_fsum.
__global__ void k_fwd(const unsigned short* x_cl, const unsigned short* T, float* Fp) {
  int b = blockIdx.x;
  int ks = blockIdx.y;
  int lane = threadIdx.x & 31, wave = threadIdx.x >> 5;
  int m0 = (wave & 3) * 16;
  int n0 = (wave >> 2) * 32;
  const int KCH = LSEQ / KSPLIT;
  size_t kbase = (size_t)ks * KCH;
  const unsigned short* arow  = x_cl + ((size_t)(b * WIDTH + m0 + (lane & 15))) * LSEQ + kbase;
  const unsigned short* brow0 = T + (size_t)(n0 + (lane & 15)) * LSEQ + kbase;
  const unsigned short* brow1 = T + (size_t)(n0 + 16 + (lane & 15)) * LSEQ + kbase;
  v8f acc0 = {}, acc1 = {};
  for (int k0 = 0; k0 < KCH; k0 += 32) {
    if (k0 + 256 < KCH) {                       // global_prefetch_b8 ahead of stream
      __builtin_prefetch(arow + k0 + 256, 0, 1);
      __builtin_prefetch(brow0 + k0 + 256, 0, 1);
      __builtin_prefetch(brow1 + k0 + 256, 0, 1);
    }
    v16bf a  = load_frag(arow, k0, lane);
    v16bf b0 = load_frag(brow0, k0, lane);
    v16bf b1 = load_frag(brow1, k0, lane);
    acc0 = WMMA_BF16(a, b0, acc0);
    acc1 = WMMA_BF16(a, b1, acc1);
  }
#pragma unroll
  for (int j = 0; j < 8; ++j) {
    int row = m0 + j + ((lane >> 4) << 3);
    size_t base = ((size_t)ks * BATCH * WIDTH + b * WIDTH + row) * 64;
    Fp[base + n0 + (lane & 15)]      = acc0[j];
    Fp[base + n0 + 16 + (lane & 15)] = acc1[j];
  }
}

// deterministic fixed-order reduction of the KSPLIT partials
__global__ void k_fsum(const float* Fp, float* F) {
  int i = blockIdx.x * blockDim.x + threadIdx.x;
  if (i >= BATCH * WIDTH * 64) return;
  float s = 0.f;
#pragma unroll
  for (int p = 0; p < KSPLIT; ++p) s += Fp[(size_t)p * BATCH * WIDTH * 64 + i];
  F[i] = s;
}

// ---------------------------------------------------------------- mode mixing
// Wc[b](64 x 128) = [ c_m*Or | c_m*Oi | pw_w ]  (bf16), c_0=1/L, c_m=2/L
__global__ void k_mix(const float* F, const float* spec_w, const float* pw_w,
                      unsigned short* Wc, int layer) {
  int idx = blockIdx.x * blockDim.x + threadIdx.x;
  if (idx >= BATCH * WIDTH * MODES) return;
  int m = idx & 31;
  int o = (idx >> 5) & 63;
  int b = idx >> 11;
  const float* Fb = F + b * WIDTH * 64;
  const float* sw = spec_w + (size_t)layer * WIDTH * WIDTH * MODES * 2;
  float orr = 0.f, oii = 0.f;
  for (int i = 0; i < WIDTH; ++i) {
    float xr = Fb[i * 64 + m];
    float xi = Fb[i * 64 + 32 + m];
    const float* w = sw + ((size_t)(i * WIDTH + o) * MODES + m) * 2;
    orr += xr * w[0] - xi * w[1];
    oii += xr * w[1] + xi * w[0];
  }
  float cm = (m == 0) ? (1.0f / (float)LSEQ) : (2.0f / (float)LSEQ);
  unsigned short* row = Wc + (size_t)(b * WIDTH + o) * 128;
  row[m]      = f2bf(cm * orr);
  row[32 + m] = f2bf(cm * oii);
  const float* pw = pw_w + ((size_t)layer * WIDTH + o) * WIDTH;
  row[64 + m] = f2bf(pw[m]);
  row[96 + m] = f2bf(pw[m + 32]);
}

// ---------------------------------------------------------------- fused iDFT + pointwise (WMMA)
// h[b](64 x 4096) = Wc[b](64 x 128) @ [cos; -sin; x[b]](128 x 4096) + pw_b
// B tiles (basis 16KB + x 16KB per 128-column chunk) are wave-invariant in M:
// one TDM tensor_load_to_lds per tile (wave 0 issues, TENSORcnt waits),
// async-to-LDS fallback if the builtin is unavailable.
__global__ void k_hmat(const unsigned short* Wc, const unsigned short* Tt,
                       const unsigned short* x_lc, const float* pw_b,
                       unsigned short* h, int layer) {
  __shared__ unsigned short ldsT[128 * 64];     // 16 KB: basis rows for 128 cols
  __shared__ unsigned short ldsX[128 * 64];     // 16 KB: x columns
  int b = blockIdx.y;
  int nchunk = blockIdx.x;
  int tid = threadIdx.x;
  int lane = tid & 31, wave = tid >> 5;

  const char* gT = (const char*)(Tt + (size_t)nchunk * 128 * 64);
  const char* gX = (const char*)(x_lc + ((size_t)b * LSEQ + nchunk * 128) * 64);
#if TDM_AVAILABLE
  if (wave == 0) {
    tdm_copy_dwords(gT, (void*)ldsT, 4096);     // 16 KB each, one D# per tile
    tdm_copy_dwords(gX, (void*)ldsX, 4096);
    __builtin_amdgcn_s_wait_tensorcnt(0);
  }
  __syncthreads();
#else
#pragma unroll
  for (int it = 0; it < 4; ++it) {
    unsigned off = (unsigned)tid * 16u + (unsigned)it * 4096u;
    async_copy16(gT + off, (char*)ldsT + off);
    async_copy16(gX + off, (char*)ldsX + off);
  }
  async_wait_all();
  __syncthreads();
#endif

  int m0 = (wave & 3) * 16;
  int nhalf = (wave >> 2) * 64;                 // local column base (0 or 64)
  const unsigned short* arow = Wc + (size_t)(b * WIDTH + m0 + (lane & 15)) * 128;
  v8f acc[4] = {};
#pragma unroll
  for (int k0 = 0; k0 < 128; k0 += 32) {
    v16bf a = load_frag(arow, k0, lane);
#pragma unroll
    for (int s = 0; s < 4; ++s) {
      int lc = nhalf + s * 16 + (lane & 15);    // 0..127 within chunk
      const unsigned short* brow = (k0 < 64) ? (ldsT + lc * 64) : (ldsX + lc * 64);
      int kk = (k0 < 64) ? k0 : (k0 - 64);
      v16bf bv = load_frag(brow, kk, lane);
      acc[s] = WMMA_BF16(a, bv, acc[s]);
    }
  }
#pragma unroll
  for (int j = 0; j < 8; ++j) {
    int row = m0 + j + ((lane >> 4) << 3);
    float bias = pw_b[layer * WIDTH + row];     // hoisted: one load per row
#pragma unroll
    for (int s = 0; s < 4; ++s) {
      int col = nchunk * 128 + nhalf + s * 16 + (lane & 15);
      h[((size_t)b * WIDTH + row) * LSEQ + col] = f2bf(acc[s][j] + bias);
    }
  }
}

// ---------------------------------------------------------------- instance-norm stats
__global__ void k_stats(const unsigned short* h, float* stats) {
  int bc = blockIdx.x;
  const unsigned short* row = h + (size_t)bc * LSEQ;
  float s = 0.f, sq = 0.f;
  for (int i = threadIdx.x; i < LSEQ; i += blockDim.x) {
    float v = bf2f(row[i]);
    s += v; sq += v * v;
  }
  __shared__ float ls[256], lq[256];
  ls[threadIdx.x] = s; lq[threadIdx.x] = sq;
  __syncthreads();
  for (int st = 128; st > 0; st >>= 1) {
    if ((int)threadIdx.x < st) {
      ls[threadIdx.x] += ls[threadIdx.x + st];
      lq[threadIdx.x] += lq[threadIdx.x + st];
    }
    __syncthreads();
  }
  if (threadIdx.x == 0) {
    float mean = ls[0] / (float)LSEQ;
    float var = lq[0] / (float)LSEQ - mean * mean;
    stats[bc * 2] = mean;
    stats[bc * 2 + 1] = rsqrtf(fmaxf(var, 0.f) + 1e-5f);
  }
}

// ---------------------------------------------------------------- normalize + gelu
__global__ void k_norm(const unsigned short* h, const float* stats,
                       unsigned short* x_cl, unsigned short* x_lc) {
  size_t idx = (size_t)blockIdx.x * blockDim.x + threadIdx.x;
  if (idx >= (size_t)BATCH * WIDTH * LSEQ) return;
  int l = (int)(idx & (LSEQ - 1));
  int c = (int)((idx >> 12) & (WIDTH - 1));
  int b = (int)(idx >> 18);
  int bc = b * WIDTH + c;
  float v = (bf2f(h[idx]) - stats[bc * 2]) * stats[bc * 2 + 1];
  unsigned short bv = f2bf(gelu_exact(v));
  x_cl[idx] = bv;
  x_lc[((size_t)b * LSEQ + l) * WIDTH + c] = bv;
}

// ---------------------------------------------------------------- projection (WMMA + fused reduce)
// P = proj_w1(128 x 64) @ x[b], gelu, then out[l] = proj_w2 . P[:,l] + b2
// B tile (64 cols x K=64 = 8KB) identical for all 8 waves -> async LDS stage.
__global__ void k_proj(const unsigned short* pw1, const unsigned short* x_lc,
                       const float* b1, const float* w2, const float* b2,
                       float* out) {
  __shared__ unsigned short ldsX[64 * WIDTH];   // 8 KB
  int b = blockIdx.y;
  int nblk = blockIdx.x;
  int tid = threadIdx.x;
  int lane = tid & 31, wave = tid >> 5;

  const char* gX = (const char*)(x_lc + ((size_t)b * LSEQ + nblk * 64) * WIDTH);
#pragma unroll
  for (int it = 0; it < 2; ++it) {
    unsigned off = (unsigned)tid * 16u + (unsigned)it * 4096u;
    async_copy16(gX + off, (char*)ldsX + off);
  }
  async_wait_all();
  __syncthreads();

  int m0 = wave * 16;
  const unsigned short* arow = pw1 + (m0 + (lane & 15)) * WIDTH;
  v8f acc[4] = {};
#pragma unroll
  for (int k0 = 0; k0 < WIDTH; k0 += 32) {
    v16bf a = load_frag(arow, k0, lane);
#pragma unroll
    for (int s = 0; s < 4; ++s) {
      const unsigned short* brow = ldsX + (s * 16 + (lane & 15)) * WIDTH;
      v16bf bv = load_frag(brow, k0, lane);
      acc[s] = WMMA_BF16(a, bv, acc[s]);
    }
  }
  __shared__ float red[8][64][2];
  float p[4] = {0.f, 0.f, 0.f, 0.f};
#pragma unroll
  for (int j = 0; j < 8; ++j) {
    int row = m0 + j + ((lane >> 4) << 3);
    float bb = b1[row];                         // hoisted across the 4 subtiles
    float ww = w2[row];
#pragma unroll
    for (int s = 0; s < 4; ++s) p[s] += ww * gelu_exact(acc[s][j] + bb);
  }
#pragma unroll
  for (int s = 0; s < 4; ++s)
    red[wave][s * 16 + (lane & 15)][lane >> 4] = p[s];
  __syncthreads();
  if (threadIdx.x < 64) {
    float sum = b2[0];
    for (int w = 0; w < 8; ++w) sum += red[w][threadIdx.x][0] + red[w][threadIdx.x][1];
    out[(size_t)b * LSEQ + nblk * 64 + threadIdx.x] = sum;
  }
}

extern "C" void kernel_launch(void* const* d_in, const int* in_sizes, int n_in,
                              void* d_out, int out_size, void* d_ws, size_t ws_size,
                              hipStream_t stream) {
  const float* z      = (const float*)d_in[0];
  const float* ip     = (const float*)d_in[1];
  const float* lift_w = (const float*)d_in[2];
  const float* lift_b = (const float*)d_in[3];
  const float* iw1    = (const float*)d_in[4];
  const float* ib1    = (const float*)d_in[5];
  const float* iw2    = (const float*)d_in[6];
  const float* ib2    = (const float*)d_in[7];
  const float* spec_w = (const float*)d_in[8];
  const float* pw_w   = (const float*)d_in[9];
  const float* pw_b   = (const float*)d_in[10];
  const float* pj_w1  = (const float*)d_in[11];
  const float* pj_b1  = (const float*)d_in[12];
  const float* pj_w2  = (const float*)d_in[13];
  const float* pj_b2  = (const float*)d_in[14];
  float* out = (float*)d_out;

  char* ws = (char*)d_ws;
  size_t off = 0;
  auto walloc = [&](size_t bytes) {
    size_t o = off;
    off += (bytes + 255) & ~(size_t)255;
    return o;
  };
  unsigned short* T    = (unsigned short*)(ws + walloc(64UL * 4096 * 2));
  unsigned short* Tt   = (unsigned short*)(ws + walloc(4096UL * 64 * 2));
  unsigned short* zbf  = (unsigned short*)(ws + walloc(128UL * 128 * 2));
  unsigned short* lwT  = (unsigned short*)(ws + walloc(4096UL * 128 * 2));
  unsigned short* p1bf = (unsigned short*)(ws + walloc(128UL * 64 * 2));
  float* u     = (float*)(ws + walloc(128UL * 4096 * 4));
  float* cond  = (float*)(ws + walloc(128UL * 64 * 4));
  float* Fp    = (float*)(ws + walloc((size_t)KSPLIT * 128 * 64 * 64 * 4));
  float* F     = (float*)(ws + walloc(128UL * 64 * 64 * 4));
  unsigned short* Wc = (unsigned short*)(ws + walloc(128UL * 64 * 128 * 2));
  float* stats = (float*)(ws + walloc(128UL * 64 * 2 * 4));
  unsigned short* x_cl = (unsigned short*)(ws + walloc(128UL * 64 * 4096 * 2));
  unsigned short* x_lc = (unsigned short*)(ws + walloc(128UL * 64 * 4096 * 2));
  unsigned short* h    = (unsigned short*)(ws + walloc(128UL * 64 * 4096 * 2));

  k_basis<<<dim3((64 * 4096) / 256), 256, 0, stream>>>(T, Tt);
  k_cvt<<<dim3((128 * 128) / 256), 256, 0, stream>>>(z, zbf, 128 * 128);
  k_cvt_liftw<<<dim3((128 * 4096) / 256), 256, 0, stream>>>(lift_w, lwT);
  k_cvt<<<dim3((128 * 64) / 256), 256, 0, stream>>>(pj_w1, p1bf, 128 * 64);
  k_lift<<<dim3(64), 256, 0, stream>>>(zbf, lwT, lift_b, u);
  k_cond<<<dim3(1), 128, 0, stream>>>(ip, iw1, ib1, iw2, ib2, cond);
  k_interp<<<dim3((128 * 64 * 4096) / 256), 256, 0, stream>>>(u, cond, x_cl, x_lc);

  for (int layer = 0; layer < 4; ++layer) {
    k_fwd<<<dim3(128, KSPLIT), 256, 0, stream>>>(x_cl, T, Fp);
    k_fsum<<<dim3((128 * 64 * 64) / 256), 256, 0, stream>>>(Fp, F);
    k_mix<<<dim3((128 * 64 * 32) / 256), 256, 0, stream>>>(F, spec_w, pw_w, Wc, layer);
    k_hmat<<<dim3(32, 128), 256, 0, stream>>>(Wc, Tt, x_lc, pw_b, h, layer);
    k_stats<<<dim3(128 * 64), 256, 0, stream>>>(h, stats);
    k_norm<<<dim3((128 * 64 * 4096) / 256), 256, 0, stream>>>(h, stats, x_cl, x_lc);
  }

  k_proj<<<dim3(64, 128), 256, 0, stream>>>(p1bf, x_lc, pj_b1, pj_w2, pj_b2, out);
}
